// LexComponents_61108794687737
// MI455X (gfx1250) — matrix operand order repaired and checked by reading, not compiled, not run on hardware
//
#include <hip/hip_runtime.h>
#include <hip/hip_bf16.h>
#include <math.h>
#include <stdint.h>

// ---------------------------------------------------------------------------
// MI455X (gfx1250) GAT-conv + GRU + LayerNorm, wave32 WMMA implementation.
// - all five GEMMs on v_wmma_f32_16x16x32_f16 (f16 A/B, f32 accum)
// - streamed activations in f16 (bandwidth-bound: ~134 GFLOP vs ~1 GB traffic)
// - edge gather + GRU tile copy via GLOBAL_LOAD_ASYNC_TO_LDS (ASYNCcnt path)
// - W1 weight staging via Tensor Data Mover (tensor_load_to_lds, TENSORcnt)
// ---------------------------------------------------------------------------

typedef _Float16 half_t;
typedef __attribute__((ext_vector_type(16))) _Float16 v16h;
typedef __attribute__((ext_vector_type(8)))  _Float16 v8h;
typedef __attribute__((ext_vector_type(8)))  float    v8f;
typedef __attribute__((ext_vector_type(4)))  int      v4i;
typedef __attribute__((ext_vector_type(4)))  uint32_t v4u;
typedef __attribute__((ext_vector_type(8)))  uint32_t v8u;

#define NN  262144
#define EE  1048576
#define DINF 64
#define HF  128
#define DEF 32

// ---- async global->LDS copy (ASYNCcnt), guarded so compile never breaks ----
#if __has_builtin(__builtin_amdgcn_global_load_async_to_lds_b128)
#define HAVE_ASYNC_LDS 1
// builtin signature (per hipcc diagnostic): (int4 AS1*, int4 AS3*, Ii, Ii)
__device__ inline void copy16_g2l(const void* g, void* l) {
  __builtin_amdgcn_global_load_async_to_lds_b128(
      (v4i __attribute__((address_space(1)))*)(uintptr_t)g,
      (v4i __attribute__((address_space(3)))*)(uint32_t)(uintptr_t)l,
      0, 0);
}
__device__ inline void async_wait0() {
#if __has_builtin(__builtin_amdgcn_s_wait_asynccnt)
  __builtin_amdgcn_s_wait_asynccnt(0);
#else
  asm volatile("s_wait_asynccnt 0x0" ::: "memory");
#endif
}
#else
#define HAVE_ASYNC_LDS 0
__device__ inline void copy16_g2l(const void* g, void* l) {
  *(v8h*)l = *(const v8h*)g;
}
__device__ inline void async_wait0() {}
#endif

// ---- WMMA fragment helpers (layouts per CDNA5 ISA 7.12.2, wave32) ----------

// A 16x32 f16: lane l holds row M=l&15; elems 0..7 -> K=hb+0..7,
// elems 8..15 -> K=16+hb+0..7, hb=(l>>4)*8. Two 16B loads.
__device__ inline v16h load_frag_a(const half_t* base, int ld, int lane) {
  int row = lane & 15;
  int hb  = (lane >> 4) * 8;
  const half_t* p = base + row * ld + hb;
  v8h lo = *(const v8h*)(p);
  v8h hi = *(const v8h*)(p + 16);
  return __builtin_shufflevector(lo, hi, 0,1,2,3,4,5,6,7,8,9,10,11,12,13,14,15);
}

// B 32x16 f16 from row-major W[N][K]: lane l holds column N=l&15;
// elems 0..7 -> K=kb+0..7, elems 8..15 -> K=16+kb+0..7, kb=(l>>4)*8.
__device__ inline v16h load_frag_b(const half_t* base, int ld, int lane) {
  int row = lane & 15;
  int kb  = (lane >> 4) * 8;
  const half_t* p = base + row * ld + kb;
  v8h lo = *(const v8h*)(p);
  v8h hi = *(const v8h*)(p + 16);
  return __builtin_shufflevector(lo, hi, 0,1,2,3,4,5,6,7,8,9,10,11,12,13,14,15);
}

__device__ inline v8f wmma16(v16h a, v16h b, v8f c) {
  return __builtin_amdgcn_wmma_f32_16x16x32_f16(false, a, false, b, (short)0, c,
                                                false, false);
}

__device__ inline float lrelu(float v)    { return v > 0.f ? v : 0.01f * v; }
__device__ inline float sigmoidf_(float v){ return 1.f / (1.f + __expf(-v)); }

// ---------------------------------------------------------------------------
__global__ void k_cvt(const float* __restrict__ in, half_t* __restrict__ out, int n) {
  int i = blockIdx.x * 256 + threadIdx.x;
  if (i < n) out[i] = (half_t)in[i];
}

__global__ void k_init(float* __restrict__ agg, float* __restrict__ denom) {
  size_t i = (size_t)blockIdx.x * 256 + threadIdx.x;
  if (i < (size_t)NN * HF) agg[i] = 0.f;
  if (i < NN) denom[i] = 0.f;
}

// ---------------------------------------------------------------------------
// k_enter: x16 = f16(leaky_relu(x @ W_enter^T + b)); one wave per 16-node tile
// ---------------------------------------------------------------------------
__global__ __launch_bounds__(256) void k_enter(
    const float* __restrict__ x, const half_t* __restrict__ We16,
    const float* __restrict__ be, half_t* __restrict__ x16) {
  __shared__ __align__(16) half_t xt[8][16][72];
  int wave = threadIdx.x >> 5, lane = threadIdx.x & 31;
  int m0 = (blockIdx.x * 8 + wave) * 16;

  const float4* src = (const float4*)(x + (size_t)m0 * DINF);
  for (int i = lane; i < 16 * 16; i += 32) {     // 16 rows x 16 float4
    int r = i >> 4, c4 = i & 15;
    float4 v = src[r * 16 + c4];
    half_t* d = &xt[wave][r][c4 * 4];
    d[0] = (half_t)v.x; d[1] = (half_t)v.y; d[2] = (half_t)v.z; d[3] = (half_t)v.w;
  }
  __syncthreads();

  int lh = lane >> 4, col = lane & 15;
  for (int nt = 0; nt < 8; ++nt) {
    v8f acc = {};
#pragma unroll
    for (int kb = 0; kb < 2; ++kb) {
      v16h a = load_frag_a(&xt[wave][0][kb * 32], 72, lane);
      v16h b = load_frag_b(We16 + (size_t)(nt * 16) * DINF + kb * 32, DINF, lane);
      acc = wmma16(a, b, acc);
    }
    float bias = be[nt * 16 + col];
#pragma unroll
    for (int r = 0; r < 8; ++r) {
      float v = lrelu(acc[r] + bias);
      x16[(size_t)(m0 + r + 8 * lh) * HF + nt * 16 + col] = (half_t)v;
    }
  }
}

// ---------------------------------------------------------------------------
// k_msg: per 16-edge tile (one wave, 2 waves/block):
//   TDM stages W1 (128x160 f16) into LDS once per block
//   async-gather x16[src] + edge_attr -> LDS [16][160]
//   m    = leaky(lin1 tile)  (WMMA 8x5, B from LDS)  -> LDS f16
//   msg0 = m @ W2^T          (WMMA 8x4, B from L2)   -> global f16
//   logit = leaky(m.att_l + x16[dst].att_r); exp -> ealpha, atomic denom
// ---------------------------------------------------------------------------
__global__ __launch_bounds__(64) void k_msg(
    const half_t* __restrict__ x16, const half_t* __restrict__ ea16,
    const half_t* __restrict__ W1h, const half_t* __restrict__ W2h,
    const float* __restrict__ att_l, const float* __restrict__ att_r,
    const int* __restrict__ src_idx, const int* __restrict__ dst_idx,
    half_t* __restrict__ msg0, float* __restrict__ ealpha,
    float* __restrict__ denom) {
  __shared__ __align__(16) half_t w1s[HF * 160];   // 40KB, TDM-staged
  __shared__ __align__(16) half_t xe[2][16][168];
  __shared__ __align__(16) half_t mt[2][16][136];
  __shared__ float arow[2][16];
  int wave = threadIdx.x >> 5, lane = threadIdx.x & 31;
  int e0 = (blockIdx.x * 2 + wave) * 16;

  if (wave == 0) {
    // Tensor DMA descriptor (ISA 8.3/8.4): 2D tensor, data_size=2B,
    // tensor/tile 160x128, stride 160; dest = w1s.
    uint64_t ga = (uint64_t)(uintptr_t)W1h;
    uint32_t la = (uint32_t)(uintptr_t)&w1s[0];
    v4u g0;
    g0[0] = 1u;                                        // count=1, no gather
    g0[1] = la;                                        // lds_addr
    g0[2] = (uint32_t)ga;                              // global_addr[31:0]
    g0[3] = (uint32_t)((ga >> 32) & 0x01FFFFFFu) | 0x80000000u; // addr[56:32]|type=2
    v8u g1;
    g1[0] = 0x00010000u;          // workgroup_mask=0, data_size=1 (2 bytes)
    g1[1] = (160u & 0xFFFFu) << 16;                    // tensor_dim0[15:0]
    g1[2] = (160u >> 16) | ((128u & 0xFFFFu) << 16);   // dim0[31:16] | dim1[15:0]
    g1[3] = (128u >> 16) | (160u << 16);               // dim1[31:16] | tile_dim0
    g1[4] = 128u;                                      // tile_dim1 (tile_dim2=0)
    g1[5] = 160u;                                      // tensor_dim0_stride[31:0]
    g1[6] = 0u;                                        // stride0[47:32]|stride1[15:0]
    g1[7] = 0u;
    asm volatile("tensor_load_to_lds %0, %1" :: "s"(g0), "s"(g1) : "memory");
    __builtin_amdgcn_s_wait_tensorcnt(0);
  }

  // gather: row = [x16[src[e]] (128h) | edge_attr16[e] (32h)], 20 v8h segs/row
  for (int i = lane; i < 16 * 20; i += 32) {
    int r = i / 20, s = i % 20;
    const half_t* g = (s < 16)
        ? x16 + (size_t)src_idx[e0 + r] * HF + s * 8
        : ea16 + (size_t)(e0 + r) * DEF + (s - 16) * 8;
    copy16_g2l(g, &xe[wave][r][s * 8]);
  }
  async_wait0();
  __syncthreads();

  int lh = lane >> 4, col = lane & 15;
  float part[8] = {};
  for (int nt = 0; nt < 8; ++nt) {
    v8f acc = {};
#pragma unroll
    for (int kb = 0; kb < 5; ++kb) {
      v16h a = load_frag_a(&xe[wave][0][kb * 32], 168, lane);
      v16h b = load_frag_b(&w1s[(size_t)(nt * 16) * 160 + kb * 32], 160, lane);
      acc = wmma16(a, b, acc);
    }
    float al = att_l[nt * 16 + col];
#pragma unroll
    for (int r = 0; r < 8; ++r) {
      float v = lrelu(acc[r]);
      part[r] += v * al;                       // partial of m . att_l
      mt[wave][r + 8 * lh][nt * 16 + col] = (half_t)v;
    }
  }
#pragma unroll
  for (int r = 0; r < 8; ++r)
    for (int m = 1; m < 16; m <<= 1) part[r] += __shfl_xor(part[r], m, 32);
  if (col == 0) {
#pragma unroll
    for (int r = 0; r < 8; ++r) arow[wave][r + 8 * lh] = part[r];
  }
  __syncthreads();

  for (int nt = 0; nt < 8; ++nt) {             // msg0 = m @ W2^T (unscaled)
    v8f acc = {};
#pragma unroll
    for (int kb = 0; kb < 4; ++kb) {
      v16h a = load_frag_a(&mt[wave][0][kb * 32], 136, lane);
      v16h b = load_frag_b(W2h + (size_t)(nt * 16) * HF + kb * 32, HF, lane);
      acc = wmma16(a, b, acc);
    }
#pragma unroll
    for (int r = 0; r < 8; ++r)
      msg0[(size_t)(e0 + r + 8 * lh) * HF + nt * 16 + col] = (half_t)acc[r];
  }

  if (lane < 16) {                             // attention logits, 1 edge/lane
    int e = e0 + lane;
    int d = dst_idx[e];
    const half_t* xd = x16 + (size_t)d * HF;
    float dot = 0.f;
#pragma unroll 4
    for (int c = 0; c < HF; ++c) dot += (float)xd[c] * att_r[c];
    float ex = __expf(lrelu(arow[wave][lane] + dot));
    ealpha[e] = ex;
    unsafeAtomicAdd(&denom[d], ex);
  }
}

// ---------------------------------------------------------------------------
// k_scatter: agg[dst] += msg0 * (ealpha / denom[dst])    (f32 atomics)
// ---------------------------------------------------------------------------
__global__ __launch_bounds__(256) void k_scatter(
    const half_t* __restrict__ msg0, const float* __restrict__ ealpha,
    const float* __restrict__ denom, const int* __restrict__ dst_idx,
    float* __restrict__ agg) {
  size_t i = (size_t)blockIdx.x * 256 + threadIdx.x;   // over E*H
  int e = (int)(i >> 7), c = (int)(i & 127);
  __builtin_prefetch(msg0 + i + 32768, 0, 0);          // global_prefetch_b8
  int d = dst_idx[e];
  float a = ealpha[e] / denom[d];
  unsafeAtomicAdd(&agg[(size_t)d * HF + c], (float)msg0[i] * a);
}

// ---------------------------------------------------------------------------
// k_gru: h = elu(agg+b_conv); gi = h@W_ih^T+b_ih; gh = x@W_hh^T+b_hh;
// GRU gates, relu, layernorm. 6 WMMA accumulators per 16-channel tile.
// ---------------------------------------------------------------------------
__global__ __launch_bounds__(64) void k_gru(
    const float* __restrict__ agg, const float* __restrict__ b_conv,
    const half_t* __restrict__ x16, const half_t* __restrict__ Wih,
    const half_t* __restrict__ Whh, const float* __restrict__ b_ih,
    const float* __restrict__ b_hh, float* __restrict__ out) {
  __shared__ __align__(16) half_t ht[2][16][136];
  __shared__ __align__(16) half_t xt[2][16][136];
  __shared__ float yt[2][16][132];
  int wave = threadIdx.x >> 5, lane = threadIdx.x & 31;
  int m0 = (blockIdx.x * 2 + wave) * 16;

  for (int i = lane; i < 16 * 16; i += 32)     // x tile: pure async copy
    copy16_g2l(x16 + (size_t)(m0 + (i >> 4)) * HF + (i & 15) * 8,
               &xt[wave][i >> 4][(i & 15) * 8]);
  for (int i = lane; i < 16 * HF; i += 32) {   // h tile: elu transform
    int r = i >> 7, c = i & 127;
    float g = agg[(size_t)(m0 + r) * HF + c] + b_conv[c];
    ht[wave][r][c] = (half_t)(g > 0.f ? g : (__expf(g) - 1.f));
  }
  async_wait0();
  __syncthreads();

  int lh = lane >> 4, col = lane & 15;
  for (int t = 0; t < 8; ++t) {
    v8f ir = {}, iz = {}, in_ = {}, hr = {}, hz = {}, hn = {};
#pragma unroll
    for (int kb = 0; kb < 4; ++kb) {
      v16h ah = load_frag_a(&ht[wave][0][kb * 32], 136, lane);
      v16h ax = load_frag_a(&xt[wave][0][kb * 32], 136, lane);
      v16h b;
      b = load_frag_b(Wih + (size_t)((t     ) * 16) * HF + kb * 32, HF, lane); ir  = wmma16(ah, b, ir);
      b = load_frag_b(Wih + (size_t)((t +  8) * 16) * HF + kb * 32, HF, lane); iz  = wmma16(ah, b, iz);
      b = load_frag_b(Wih + (size_t)((t + 16) * 16) * HF + kb * 32, HF, lane); in_ = wmma16(ah, b, in_);
      b = load_frag_b(Whh + (size_t)((t     ) * 16) * HF + kb * 32, HF, lane); hr  = wmma16(ax, b, hr);
      b = load_frag_b(Whh + (size_t)((t +  8) * 16) * HF + kb * 32, HF, lane); hz  = wmma16(ax, b, hz);
      b = load_frag_b(Whh + (size_t)((t + 16) * 16) * HF + kb * 32, HF, lane); hn  = wmma16(ax, b, hn);
    }
    int c = t * 16 + col;
    float bir = b_ih[c], biz = b_ih[c + HF], bin = b_ih[c + 2 * HF];
    float bhr = b_hh[c], bhz = b_hh[c + HF], bhn = b_hh[c + 2 * HF];
#pragma unroll
    for (int r = 0; r < 8; ++r) {
      int M = r + 8 * lh;
      float rg = sigmoidf_(ir[r] + bir + hr[r] + bhr);
      float zg = sigmoidf_(iz[r] + biz + hz[r] + bhz);
      float nc = tanhf(in_[r] + bin + rg * (hn[r] + bhn));
      float xv = (float)xt[wave][M][c];
      float y  = (1.f - zg) * nc + zg * xv;
      yt[wave][M][c] = y > 0.f ? y : 0.f;               // relu
    }
  }
  __syncthreads();

  if (lane < 16) {                                       // layernorm per node
    int M = lane;
    float mu = 0.f;
#pragma unroll 4
    for (int c = 0; c < HF; ++c) mu += yt[wave][M][c];
    mu *= (1.f / HF);
    float var = 0.f;
#pragma unroll 4
    for (int c = 0; c < HF; ++c) { float d = yt[wave][M][c] - mu; var += d * d; }
    float inv = rsqrtf(var * (1.f / HF) + 1e-5f);
#pragma unroll 4
    for (int c = 0; c < HF; ++c)
      out[(size_t)(m0 + M) * HF + c] = (yt[wave][M][c] - mu) * inv;
  }
}

// ---------------------------------------------------------------------------
// host launcher
// ---------------------------------------------------------------------------
static constexpr size_t align_up(size_t x) { return (x + 255) & ~(size_t)255; }

extern "C" void kernel_launch(void* const* d_in, const int* in_sizes, int n_in,
                              void* d_out, int out_size, void* d_ws, size_t ws_size,
                              hipStream_t stream) {
  const float* x      = (const float*)d_in[0];
  const float* eattr  = (const float*)d_in[1];
  const float* We     = (const float*)d_in[2];
  const float* be     = (const float*)d_in[3];
  const float* W1     = (const float*)d_in[4];
  const float* att_l  = (const float*)d_in[5];
  const float* att_r  = (const float*)d_in[6];
  const float* W2     = (const float*)d_in[7];
  const float* b_conv = (const float*)d_in[8];
  const float* W_ih   = (const float*)d_in[9];
  const float* W_hh   = (const float*)d_in[10];
  const float* b_ih   = (const float*)d_in[11];
  const float* b_hh   = (const float*)d_in[12];
  const int*   eidx   = (const int*)d_in[13];
  const int* src_idx = eidx;
  const int* dst_idx = eidx + EE;
  float* outp = (float*)d_out;

  // workspace carve-up (~517 MiB total)
  char* w = (char*)d_ws;
  size_t o = 0;
  half_t* We16  = (half_t*)(w + o); o += align_up((size_t)HF * DINF * 2);
  half_t* W1h   = (half_t*)(w + o); o += align_up((size_t)HF * 160 * 2);
  half_t* W2h   = (half_t*)(w + o); o += align_up((size_t)HF * HF * 2);
  half_t* Wih16 = (half_t*)(w + o); o += align_up((size_t)3 * HF * HF * 2);
  half_t* Whh16 = (half_t*)(w + o); o += align_up((size_t)3 * HF * HF * 2);
  half_t* ea16  = (half_t*)(w + o); o += align_up((size_t)EE * DEF * 2);
  half_t* x16   = (half_t*)(w + o); o += align_up((size_t)NN * HF * 2);
  half_t* msg0  = (half_t*)(w + o); o += align_up((size_t)EE * HF * 2);
  float*  ealph = (float*) (w + o); o += align_up((size_t)EE * 4);
  float*  denom = (float*) (w + o); o += align_up((size_t)NN * 4);
  float*  agg   = (float*) (w + o); o += align_up((size_t)NN * HF * 4);
  (void)in_sizes; (void)n_in; (void)out_size; (void)ws_size;

  auto cvt = [&](const float* in, half_t* op, int n) {
    k_cvt<<<(n + 255) / 256, 256, 0, stream>>>(in, op, n);
  };
  cvt(We,    We16,  HF * DINF);
  cvt(W1,    W1h,   HF * 160);
  cvt(W2,    W2h,   HF * HF);
  cvt(W_ih,  Wih16, 3 * HF * HF);
  cvt(W_hh,  Whh16, 3 * HF * HF);
  cvt(eattr, ea16,  EE * DEF);

  k_init<<<(NN * HF) / 256, 256, 0, stream>>>(agg, denom);

  k_enter<<<NN / (16 * 8), 256, 0, stream>>>(x, We16, be, x16);

  k_msg<<<EE / (16 * 2), 64, 0, stream>>>(x16, ea16, W1h, W2h, att_l, att_r,
                                          src_idx, dst_idx, msg0, ealph, denom);

  k_scatter<<<((size_t)EE * HF) / 256, 256, 0, stream>>>(msg0, ealph, denom,
                                                         dst_idx, agg);

  k_gru<<<NN / (16 * 2), 64, 0, stream>>>(agg, b_conv, x16, Wih16, Whh16,
                                          b_ih, b_hh, outp);
}
